// NeuralPeronaMalik_35914516529892
// MI455X (gfx1250) — compile-verified, matrix operand order checked by reading, not run on hardware
//
#include <hip/hip_runtime.h>

typedef __attribute__((ext_vector_type(16))) _Float16 v16h;
typedef __attribute__((ext_vector_type(8)))  float    v8f;

#define HW    512
#define PLANE (HW*HW)
#define LAMBDA 0.1f
#define EPSV   1e-5f

__device__ __forceinline__ int iclamp(int v, int lo, int hi) {
    return v < lo ? lo : (v > hi ? hi : v);
}

// Compile-time gather offset for A-fragment element with flat K index k:
// k -> (c = k/9, ky = (k%9)/3, kx = k%3) ; offset = c*chstride + ky*rowstride + kx
__host__ __device__ constexpr int offc(int k, int chstride, int rowstride) {
    return (k / 9) * chstride + ((k % 9) / 3) * rowstride + ((k % 9) % 3);
}

// ---------------------------------------------------------------------------
// Weight packing into WMMA B-fragment layout (f32 -> f16).
// Fragment = 32 lanes x 16 halfs. lane: n = lane&15, hi = lane>>4.
// element e -> klocal = (e<8) ? hi*8+e : 8+e+hi*8 ; k = chunk*32 + klocal.
// Fragments: conv1: f=chunk*2+nt (8), conv2: f=8+chunk (9), conv3: f=17+chunk (5).
// ---------------------------------------------------------------------------
__global__ void pack_weights(const float* __restrict__ w1,
                             const float* __restrict__ w2,
                             const float* __restrict__ w3,
                             _Float16* __restrict__ bpack)
{
    const int total = 22 * 512;
    for (int idx = threadIdx.x + blockIdx.x * blockDim.x; idx < total;
         idx += blockDim.x * gridDim.x) {
        int f    = idx >> 9;
        int rem  = idx & 511;
        int lane = rem >> 4;
        int e    = rem & 15;
        int hi   = lane >> 4;
        int nl   = lane & 15;
        int klocal = (e < 8) ? (hi * 8 + e) : (8 + e + hi * 8);
        float val = 0.f;
        if (f < 8) {                       // conv1: Cin=12, Kreal=108, Cout=32
            int chunk = f >> 1, nt = f & 1;
            int k = chunk * 32 + klocal;
            int n = nt * 16 + nl;
            if (k < 108) { int c = k / 9, r9 = k % 9; val = w1[(n * 12 + c) * 9 + r9]; }
        } else if (f < 17) {               // conv2: Cin=32, K=288, Cout=16
            int chunk = f - 8;
            int k = chunk * 32 + klocal;
            int c = k / 9, r9 = k % 9;
            val = w2[(nl * 32 + c) * 9 + r9];
        } else {                           // conv3: Cin=16, Kreal=144, Cout=4
            int chunk = f - 17;
            int k = chunk * 32 + klocal;
            if (k < 144 && nl < 4) { int c = k / 9, r9 = k % 9; val = w3[(nl * 16 + c) * 9 + r9]; }
        }
        bpack[idx] = (_Float16)val;
    }
}

// ---------------------------------------------------------------------------
// One-time guidance pre-pass, fully fused per 16x16 tile:
// avg_pool3 -> dilated conv(1->16)+BN+ReLU -> conv(16->8)+sigmoid -> f16 buffer
// ---------------------------------------------------------------------------
__global__ __launch_bounds__(256)
void guidance_kernel(const float* __restrict__ x,
                     const float* __restrict__ gw1, const float* __restrict__ gb1,
                     const float* __restrict__ gamma, const float* __restrict__ beta,
                     const float* __restrict__ mean,  const float* __restrict__ var,
                     const float* __restrict__ gw2,   const float* __restrict__ gb2,
                     _Float16* __restrict__ guid)
{
    __shared__ float xt[24 * 24];       // x, zero padded, origin (oy-4, ox-4)
    __shared__ float xs[22 * 24];       // avgpool, origin (oy-3, ox-3)
    __shared__ float gs[16 * 18 * 20];  // g, 16ch, origin (oy-1, ox-1)
    __shared__ float inv_s[16], shift_s[16];

    const int tid = threadIdx.x;
    const int b = blockIdx.z, oy = blockIdx.y * 16, ox = blockIdx.x * 16;
    const float* xp = x + (size_t)b * PLANE;

    if (tid < 16) {
        float inv = gamma[tid] * rsqrtf(var[tid] + EPSV);
        inv_s[tid]   = inv;
        shift_s[tid] = beta[tid] - mean[tid] * inv;
    }
    for (int idx = tid; idx < 576; idx += 256) {
        int r = idx / 24, c = idx % 24;
        int gy = oy - 4 + r, gx = ox - 4 + c;
        float v = 0.f;
        if (gy >= 0 && gy < HW && gx >= 0 && gx < HW) v = xp[gy * HW + gx];
        xt[idx] = v;
    }
    __syncthreads();
    for (int idx = tid; idx < 484; idx += 256) {
        int r = idx / 22, c = idx % 22;
        int gy = oy - 3 + r, gx = ox - 3 + c;
        float v = 0.f;
        if (gy >= 0 && gy < HW && gx >= 0 && gx < HW) {
            float s = 0.f;
            #pragma unroll
            for (int dy = 0; dy < 3; ++dy)
                #pragma unroll
                for (int dx = 0; dx < 3; ++dx)
                    s += xt[(r + dy) * 24 + (c + dx)];
            v = s * (1.f / 9.f);
        }
        xs[r * 24 + c] = v;
    }
    __syncthreads();
    for (int idx = tid; idx < 5184; idx += 256) {   // 16ch x 18x18
        int ci = idx / 324, rem = idx % 324;
        int r = rem / 18, c = rem % 18;
        int gy = oy - 1 + r, gx = ox - 1 + c;
        float v = 0.f;
        if (gy >= 0 && gy < HW && gx >= 0 && gx < HW) {
            float acc = gb1[ci];
            #pragma unroll
            for (int ky = 0; ky < 3; ++ky)
                #pragma unroll
                for (int kx = 0; kx < 3; ++kx)
                    acc += gw1[ci * 9 + ky * 3 + kx] * xs[(r + 2 * ky) * 24 + (c + 2 * kx)];
            v = fmaxf(acc * inv_s[ci] + shift_s[ci], 0.f);
        }
        gs[(ci * 18 + r) * 20 + c] = v;
    }
    __syncthreads();
    for (int idx = tid; idx < 2048; idx += 256) {   // 8ch x 16x16
        int co = idx >> 8, p = idx & 255;
        int py = p >> 4, px = p & 15;
        float acc = gb2[co];
        for (int ci = 0; ci < 16; ++ci)
            #pragma unroll
            for (int ky = 0; ky < 3; ++ky)
                #pragma unroll
                for (int kx = 0; kx < 3; ++kx)
                    acc += gw2[(co * 16 + ci) * 9 + ky * 3 + kx] *
                           gs[(ci * 18 + py + ky) * 20 + (px + kx)];
        float s = 1.f / (1.f + __expf(-acc));
        guid[(size_t)(b * 8 + co) * PLANE + (oy + py) * HW + (ox + px)] = (_Float16)s;
    }
}

// ---------------------------------------------------------------------------
// One fully-fused Perona-Malik iteration per 16x16 output tile.
// comb(15ch,22x22, ch12-14 zero) -> WMMA conv1 -> c1(32ch,20x20) -> WMMA conv2
// -> c2(18ch,18x18, ch16-17 zero) -> WMMA conv3 -> c(4ch,16x16) -> x update.
// A-fragment gather offsets are compile-time constants (one v_cndmask on the
// lane-half bit). conv1 computes both N-tiles per job so each A fragment is
// gathered from LDS exactly once.
// ---------------------------------------------------------------------------
__global__ __launch_bounds__(256, 2)
void pm_iter_kernel(const float* __restrict__ xin, float* __restrict__ xout,
                    const _Float16* __restrict__ guid,
                    const _Float16* __restrict__ bpack,
                    const float* __restrict__ cb1, const float* __restrict__ cb2,
                    const float* __restrict__ cb3)
{
    __shared__ float    xt[24 * 24];            // edge-clamped x, origin (oy-4,ox-4)
    __shared__ _Float16 comb[15 * 22 * 24];     // origin (oy-3,ox-3); ch stride 528
    __shared__ _Float16 c1s[32 * 20 * 22];      // origin (oy-2,ox-2); ch stride 440
    __shared__ _Float16 c2s[18 * 18 * 20];      // origin (oy-1,ox-1); ch stride 360
    __shared__ float    c3x[4 * 256];           // sigmoid coefficients

    const int tid  = threadIdx.x;
    const int lane = tid & 31, wave = tid >> 5;
    const int hi   = (lane >> 4) & 1, nl = lane & 15;
    const int b = blockIdx.z, oy = blockIdx.y * 16, ox = blockIdx.x * 16;
    const float* xp = xin + (size_t)b * PLANE;

    // ---- zero K-padding channels of c2s (16,17) ----
    for (int idx = tid; idx < 2 * 360; idx += 256)
        c2s[16 * 360 + idx] = (_Float16)0.f;

    // ---- stage edge-clamped x tile (matches jnp.pad mode='edge') ----
    for (int idx = tid; idx < 576; idx += 256) {
        int r = idx / 24, c = idx % 24;
        int cy = iclamp(oy - 4 + r, 0, HW - 1);
        int cx = iclamp(ox - 4 + c, 0, HW - 1);
        xt[idx] = xp[cy * HW + cx];
    }
    __syncthreads();

    // ---- build comb: ch0-3 gradients, ch4-11 guidance, ch12-14 zero pad ----
    for (int idx = tid; idx < 15 * 484; idx += 256) {
        int c = idx / 484, rem = idx % 484;
        int r = rem / 22, cc = rem % 22;
        int gy = oy - 3 + r, gx = ox - 3 + cc;
        float v = 0.f;
        if (c < 12 && gy >= 0 && gy < HW && gx >= 0 && gx < HW) {
            if (c < 4) {
                float ctr = xt[(r + 1) * 24 + cc + 1];
                float nb;
                if (c == 0)      nb = xt[(r    ) * 24 + cc + 1];   // N
                else if (c == 1) nb = xt[(r + 2) * 24 + cc + 1];   // S
                else if (c == 2) nb = xt[(r + 1) * 24 + cc + 2];   // E
                else             nb = xt[(r + 1) * 24 + cc    ];   // W
                v = nb - ctr;
            } else {
                v = (float)guid[(size_t)(b * 8 + (c - 4)) * PLANE + gy * HW + gx];
            }
        }
        comb[(c * 22 + r) * 24 + cc] = (_Float16)v;
    }
    __syncthreads();

    // ---- conv1: 12->32, out 20x20 (=400 px, 25 M-tiles x both N-tiles), ReLU ----
    for (int mt = wave; mt < 25; mt += 8) {
        int p = mt * 16 + nl;
        int py = p / 20, px = p % 20;
        int pixbase = py * 24 + px;
        v8f acc0 = {}, acc1 = {};
        #pragma unroll
        for (int chunk = 0; chunk < 4; ++chunk) {
            v16h a;
            #pragma unroll
            for (int e = 0; e < 16; ++e) {
                const int kb = chunk * 32 + ((e < 8) ? e : (8 + e));  // hi==0 k
                const int o0 = offc(kb,     528, 24);
                const int o1 = offc(kb + 8, 528, 24);
                a[e] = comb[pixbase + (hi ? o1 : o0)];
            }
            v16h b0 = *(const v16h*)(bpack + ((chunk * 2 + 0) << 9) + (lane << 4));
            v16h b1 = *(const v16h*)(bpack + ((chunk * 2 + 1) << 9) + (lane << 4));
            acc0 = __builtin_amdgcn_wmma_f32_16x16x32_f16(
                       false, a, false, b0, (short)0, acc0, false, false);
            acc1 = __builtin_amdgcn_wmma_f32_16x16x32_f16(
                       false, a, false, b1, (short)0, acc1, false, false);
        }
        float bias0 = cb1[nl];
        float bias1 = cb1[16 + nl];
        #pragma unroll
        for (int r = 0; r < 8; ++r) {
            int pp = mt * 16 + r + 8 * hi;
            int pidx = (pp / 20) * 22 + (pp % 20);
            c1s[nl * 440 + pidx]        = (_Float16)fmaxf(acc0[r] + bias0, 0.f);
            c1s[(16 + nl) * 440 + pidx] = (_Float16)fmaxf(acc1[r] + bias1, 0.f);
        }
    }
    __syncthreads();

    // ---- conv2: 32->16, out 18x18 (=324 px, 21 M-tiles), K=288, ReLU ----
    // Rows p>=324 compute garbage in D but are discarded in the epilogue.
    for (int mt = wave; mt < 21; mt += 8) {
        int p = mt * 16 + nl;
        int pc = p < 324 ? p : 323;          // address-safety clamp only
        int py = pc / 18, px = pc % 18;
        int pixbase = py * 22 + px;
        v8f acc = {};
        #pragma unroll
        for (int chunk = 0; chunk < 9; ++chunk) {
            v16h a;
            #pragma unroll
            for (int e = 0; e < 16; ++e) {
                const int kb = chunk * 32 + ((e < 8) ? e : (8 + e));
                const int o0 = offc(kb,     440, 22);
                const int o1 = offc(kb + 8, 440, 22);
                a[e] = c1s[pixbase + (hi ? o1 : o0)];
            }
            v16h bf = *(const v16h*)(bpack + ((8 + chunk) << 9) + (lane << 4));
            acc = __builtin_amdgcn_wmma_f32_16x16x32_f16(
                      false, a, false, bf, (short)0, acc, false, false);
        }
        float bias = cb2[nl];
        #pragma unroll
        for (int r = 0; r < 8; ++r) {
            int pp = mt * 16 + r + 8 * hi;
            if (pp < 324) {
                float v = fmaxf(acc[r] + bias, 0.f);
                c2s[nl * 360 + (pp / 18) * 20 + (pp % 18)] = (_Float16)v;
            }
        }
    }
    __syncthreads();

    // ---- conv3: 16->4 (N padded to 16), out 16x16, K=144 (padded 160), sigmoid ----
    for (int mt = wave; mt < 16; mt += 8) {
        int p = mt * 16 + nl;
        int py = p >> 4, px = p & 15;
        int pixbase = py * 20 + px;
        v8f acc = {};
        #pragma unroll
        for (int chunk = 0; chunk < 5; ++chunk) {
            v16h a;
            #pragma unroll
            for (int e = 0; e < 16; ++e) {
                const int kb = chunk * 32 + ((e < 8) ? e : (8 + e));
                const int o0 = offc(kb,     360, 20);
                const int o1 = offc(kb + 8, 360, 20);
                a[e] = c2s[pixbase + (hi ? o1 : o0)];
            }
            v16h bf = *(const v16h*)(bpack + ((17 + chunk) << 9) + (lane << 4));
            acc = __builtin_amdgcn_wmma_f32_16x16x32_f16(
                      false, a, false, bf, (short)0, acc, false, false);
        }
        if (nl < 4) {
            float bias = cb3[nl];
            #pragma unroll
            for (int r = 0; r < 8; ++r) {
                int pp = mt * 16 + r + 8 * hi;
                float v = acc[r] + bias;
                c3x[nl * 256 + pp] = 1.f / (1.f + __expf(-v));
            }
        }
    }
    __syncthreads();

    // ---- diffusion update: x += LAM*(cN*gN + cS*gS + cE*gE + cW*gW) ----
    {
        int p = tid;
        int py = p >> 4, px = p & 15;
        float ctr = xt[(py + 4) * 24 + px + 4];
        float gN = xt[(py + 3) * 24 + px + 4] - ctr;
        float gS = xt[(py + 5) * 24 + px + 4] - ctr;
        float gE = xt[(py + 4) * 24 + px + 5] - ctr;
        float gW = xt[(py + 4) * 24 + px + 3] - ctr;
        float xn = ctr + LAMBDA * (c3x[p] * gN + c3x[256 + p] * gS +
                                   c3x[512 + p] * gE + c3x[768 + p] * gW);
        xout[(size_t)b * PLANE + (oy + py) * HW + (ox + px)] = xn;
    }
}

// ---------------------------------------------------------------------------
extern "C" void kernel_launch(void* const* d_in, const int* in_sizes, int n_in,
                              void* d_out, int out_size, void* d_ws, size_t ws_size,
                              hipStream_t stream)
{
    const float* x     = (const float*)d_in[0];
    const float* gw1   = (const float*)d_in[1];
    const float* gb1   = (const float*)d_in[2];
    const float* gamma = (const float*)d_in[3];
    const float* beta  = (const float*)d_in[4];
    const float* mean  = (const float*)d_in[5];
    const float* var   = (const float*)d_in[6];
    const float* gw2   = (const float*)d_in[7];
    const float* gb2   = (const float*)d_in[8];
    const float* cw1   = (const float*)d_in[9];
    const float* cb1   = (const float*)d_in[10];
    const float* cw2   = (const float*)d_in[11];
    const float* cb2   = (const float*)d_in[12];
    const float* cw3   = (const float*)d_in[13];
    const float* cb3   = (const float*)d_in[14];

    char* ws = (char*)d_ws;
    float*    bufA  = (float*)ws;                                    // 8 MB
    float*    bufB  = (float*)(ws + (size_t)8 * PLANE * 4);          // 8 MB
    _Float16* guid  = (_Float16*)(ws + (size_t)2 * 8 * PLANE * 4);   // 32 MB
    _Float16* bpack = (_Float16*)(ws + (size_t)2 * 8 * PLANE * 4 +
                                       (size_t)8 * 8 * PLANE * 2);   // 22.5 KB

    pack_weights<<<4, 256, 0, stream>>>(cw1, cw2, cw3, bpack);

    dim3 grid(32, 32, 8);
    guidance_kernel<<<grid, 256, 0, stream>>>(x, gw1, gb1, gamma, beta, mean, var,
                                              gw2, gb2, guid);

    const float* src = x;
    for (int it = 0; it < 10; ++it) {
        float* dst = (it == 9) ? (float*)d_out : ((it & 1) ? bufB : bufA);
        pm_iter_kernel<<<grid, 256, 0, stream>>>(src, dst, guid, bpack, cb1, cb2, cb3);
        src = dst;
    }
}
